// HPG_52226802319569
// MI455X (gfx1250) — matrix operand order repaired and checked by reading, not compiled, not run on hardware
//
#include <hip/hip_runtime.h>
#include <cstdint>

#define N_NODES 50000
#define N_EDGES 1000000
#define DIN     2000
#define DH      20
#define BN_EPS  1e-5f
#define SLOPE   0.01f

typedef _Float16 v8h  __attribute__((ext_vector_type(8)));
typedef _Float16 v16h __attribute__((ext_vector_type(16)));
typedef float    v8f  __attribute__((ext_vector_type(8)));
typedef int      v4i  __attribute__((__vector_size__(16)));

#if __has_builtin(__builtin_amdgcn_global_load_async_to_lds_b128) && \
    __has_builtin(__builtin_amdgcn_s_wait_asynccnt)
#define USE_ASYNC_LDS 1
#else
#define USE_ASYNC_LDS 0
#endif

// ---------------------------------------------------------------------------
// Weight pre-pack: B-fragment layout for v_wmma_f32_16x16x32_f16 (wave32).
// Per (kstep, ntile): 32 lanes x 16 halves; lane holds column col=nt*16+(lane&15),
// k = kstep*32 + (lane>>4)*16 + h  (16 contiguous halves -> one 32B load).
// ---------------------------------------------------------------------------
__global__ void pack_big(const float* __restrict__ W1, const float* __restrict__ W2,
                         const float* __restrict__ b1, const float* __restrict__ b2,
                         _Float16* __restrict__ Wp, float* __restrict__ biasp) {
    const int ksteps = 63, ntiles = 20;               // Kp=2016, Cout=320
    const int total  = ksteps * ntiles * 512;
    for (int q = blockIdx.x * blockDim.x + threadIdx.x; q < total;
         q += gridDim.x * blockDim.x) {
        int kstep = q / (ntiles * 512);
        int rem   = q % (ntiles * 512);
        int nt    = rem / 512;
        int lr    = rem % 512;
        int lane  = lr >> 4;
        int h     = lr & 15;
        int col   = nt * 16 + (lane & 15);
        int k     = kstep * 32 + ((lane >> 4) << 4) + h;
        float v = 0.f;
        if (k < DIN) {
            if (col < 160) v = W1[((size_t)(col / DH) * DIN + k) * DH + (col % DH)];
            else { int c = col - 160; v = W2[((size_t)(c / DH) * DIN + k) * DH + (c % DH)]; }
        }
        Wp[q] = (_Float16)v;
    }
    int t = blockIdx.x * blockDim.x + threadIdx.x;
    if (t < 320) biasp[t] = (t < 160) ? b1[t] : b2[t - 160];
}

__global__ void pack_small(const float* __restrict__ W, const float* __restrict__ b,
                           _Float16* __restrict__ Wp, float* __restrict__ biasp) {
    const int ksteps = 2, ntiles = 8, K = 2 * DH;     // Kp=64, Cout padded 80->128
    const int total  = ksteps * ntiles * 512;
    for (int q = blockIdx.x * blockDim.x + threadIdx.x; q < total;
         q += gridDim.x * blockDim.x) {
        int kstep = q / (ntiles * 512);
        int rem   = q % (ntiles * 512);
        int nt    = rem / 512;
        int lr    = rem % 512;
        int lane  = lr >> 4;
        int h     = lr & 15;
        int col   = nt * 16 + (lane & 15);
        int k     = kstep * 32 + ((lane >> 4) << 4) + h;
        float v = 0.f;
        if (k < K && col < 80)
            v = W[((size_t)(col / DH) * K + k) * DH + (col % DH)];
        Wp[q] = (_Float16)v;
    }
    int t = blockIdx.x * blockDim.x + threadIdx.x;
    if (t < 128) biasp[t] = (t < 80) ? b[t] : 0.f;
}

// ---------------------------------------------------------------------------
// WMMA GEMM: C[M x Cout] = A[M x K] @ W + bias.  Block = 8 wave32s:
// 2 M-tiles x 4 N-groups; each wave owns NTPW contiguous n-tiles (uniform, no
// exec-mask juggling).  A tile (32 rows x 32 k, f32) is double-buffered in LDS
// and staged with global_load_async_to_lds_b128 when available (one B128 per
// thread), synchronized with s_wait_asynccnt + barrier.  K-tail steps use
// guarded synchronous zero-padded stores.  Rows >= M are clamped: their
// garbage only affects C rows >= M, which are never stored.
// ---------------------------------------------------------------------------
template <int NTILES, int NTPW>
__global__ void gemm_wmma(const float* __restrict__ A, int M, int lda, int K,
                          int ksteps, int Cout,
                          const _Float16* __restrict__ Wp,
                          const float* __restrict__ biasp,
                          float* __restrict__ C) {
    static_assert(NTILES == 4 * NTPW, "8 waves = 2 M-tiles x 4 N-groups");
    __shared__ __align__(32) float As[2][32 * 32];

    const int tid       = threadIdx.x;
    const int lane      = tid & 31;
    const int wave      = tid >> 5;
    const int mtile     = wave & 1;
    const int ngroup    = wave >> 1;
    const int rowInTile = lane & 15;     // A row / C column within tile
    const int halfsel   = lane >> 4;     // K-half for A, M+8 half for C
    const int m0        = blockIdx.x * 32;

    const int srow = tid >> 3;           // staging: row 0..31
    const int skq  = (tid & 7) << 2;     // staging: k offset 0..28 step 4

    int grow = m0 + srow;
    if (grow >= M) grow = M - 1;         // clamp (see header comment)

    auto stage = [&](int kstep, int buf) {
        int k = kstep * 32 + skq;
        const float* gp = A + (size_t)grow * lda + k;
        float*       lp = &As[buf][srow * 32 + skq];
        bool full = (kstep * 32 + 32 <= K);
#if USE_ASYNC_LDS
        if (full) {
            __builtin_amdgcn_global_load_async_to_lds_b128(
                (__attribute__((address_space(1))) v4i*)gp,
                (__attribute__((address_space(3))) v4i*)lp, 0, 0);
            return;
        }
#endif
        float4 v = make_float4(0.f, 0.f, 0.f, 0.f);
        if (full) {
            v = *(const float4*)gp;
        } else {
            if (k + 0 < K) v.x = gp[0];
            if (k + 1 < K) v.y = gp[1];
            if (k + 2 < K) v.z = gp[2];
            if (k + 3 < K) v.w = gp[3];
        }
        *(float4*)lp = v;
    };

    v8f acc[NTPW];
#pragma unroll
    for (int c = 0; c < NTPW; ++c)
        acc[c] = (v8f){0.f, 0.f, 0.f, 0.f, 0.f, 0.f, 0.f, 0.f};

    stage(0, 0);                         // pipeline prologue

    for (int kstep = 0; kstep < ksteps; ++kstep) {
        const int cur = kstep & 1;
#if USE_ASYNC_LDS
        __builtin_amdgcn_s_wait_asynccnt(0);
#endif
        __syncthreads();                 // tile `cur` visible; buf `cur^1` free
        if (kstep + 1 < ksteps) stage(kstep + 1, cur ^ 1);

        // ---- A fragment, contiguous k = halfsel*16 + s as in packed B (f32 LDS -> f16 regs) ----
        const float* ab = &As[cur][(mtile * 16 + rowInTile) * 32 + halfsel * 16];
        v8f flo = *(const v8f*)ab;
        v8f fhi = *(const v8f*)(ab + 8);
        v8h alo = __builtin_convertvector(flo, v8h);
        v8h ahi = __builtin_convertvector(fhi, v8h);
        v16h a  = __builtin_shufflevector(alo, ahi,
                  0, 1, 2, 3, 4, 5, 6, 7, 8, 9, 10, 11, 12, 13, 14, 15);

        const _Float16* wb = Wp + ((size_t)kstep * NTILES + ngroup * NTPW) * 512;
        if (kstep + 1 < ksteps)
            __builtin_prefetch(wb + (size_t)NTILES * 512 + lane * 16, 0, 0);

        v16h bf[NTPW];
#pragma unroll
        for (int c = 0; c < NTPW; ++c)
            bf[c] = *(const v16h*)(wb + (size_t)c * 512 + lane * 16);
#pragma unroll
        for (int c = 0; c < NTPW; ++c)
            acc[c] = __builtin_amdgcn_wmma_f32_16x16x32_f16(
                false, a, false, bf[c], (short)0, acc[c], false, false);
    }

    // ---- epilogue: C layout -> col = lane&15, row = base + r + (lane>>4)*8 ----
#pragma unroll
    for (int c = 0; c < NTPW; ++c) {
        int col = (ngroup * NTPW + c) * 16 + rowInTile;
        if (col < Cout) {
            float bs    = biasp[col];
            int   rbase = m0 + mtile * 16 + halfsel * 8;
#pragma unroll
            for (int r = 0; r < 8; ++r) {
                int row = rbase + r;
                if (row < M) C[(size_t)row * Cout + col] = acc[c][r] + bs;
            }
        }
    }
}

// ---------------------------------------------------------------------------
// Attention / BN kernels. P layout per conv: [q | k | v | skip] at col offsets
// 0, 20, 40, 60 with leading dimension ldP.
// ---------------------------------------------------------------------------
__device__ __forceinline__ int orderFloat(float x) {
    int i = __float_as_int(x);
    return (i >= 0) ? i : (i ^ 0x7fffffff);
}
__device__ __forceinline__ float unorderFloat(int i) {
    return __int_as_float((i >= 0) ? i : (i ^ 0x7fffffff));
}

__global__ void k_attn_init(int* __restrict__ mInt, float* __restrict__ denom,
                            float* __restrict__ agg, float* __restrict__ colstats) {
    int i = blockIdx.x * blockDim.x + threadIdx.x;
    if (i < N_NODES * DH) agg[i] = 0.f;
    if (i < N_NODES) { mInt[i] = (-2147483647 - 1); denom[i] = 0.f; }
    if (i < 2 * DH) colstats[i] = 0.f;
}

__global__ void k_logits(const int* __restrict__ ei, const float* __restrict__ P,
                         int ldP, float* __restrict__ logits, int* __restrict__ mInt) {
    int e = blockIdx.x * blockDim.x + threadIdx.x;
    if (e >= N_EDGES) return;
    int src = ei[e], dst = ei[N_EDGES + e];
    const float4* q  = (const float4*)(P + (size_t)dst * ldP);
    const float4* kk = (const float4*)(P + (size_t)src * ldP + DH);
    float s = 0.f;
#pragma unroll
    for (int j = 0; j < 5; ++j) {
        float4 a = q[j], b = kk[j];
        s += a.x * b.x + a.y * b.y + a.z * b.z + a.w * b.w;
    }
    s *= 0.22360679774997896f;  // 1/sqrt(20)
    logits[e] = s;
    atomicMax(&mInt[dst], orderFloat(s));
}

__global__ void k_expsum(const int* __restrict__ ei, const float* __restrict__ P,
                         int ldP, const float* __restrict__ logits,
                         const int* __restrict__ mInt, float* __restrict__ denom,
                         float* __restrict__ agg) {
    int e = blockIdx.x * blockDim.x + threadIdx.x;
    if (e >= N_EDGES) return;
    int src = ei[e], dst = ei[N_EDGES + e];
    float m = unorderFloat(mInt[dst]);
    float w = __expf(logits[e] - m);
    atomicAdd(&denom[dst], w);
    const float* v = P + (size_t)src * ldP + 2 * DH;
    float*       a = agg + (size_t)dst * DH;
#pragma unroll
    for (int j = 0; j < DH; ++j) atomicAdd(&a[j], w * v[j]);
}

__global__ void k_finalize_stats(const float* __restrict__ P, int ldP,
                                 const float* __restrict__ denom,
                                 const float* __restrict__ agg,
                                 float* __restrict__ t, float* __restrict__ colstats) {
    __shared__ float ssum[DH], ssq[DH];
    int tid = threadIdx.x;
    if (tid < DH) ssum[tid] = 0.f;
    else if (tid < 2 * DH) ssq[tid - DH] = 0.f;
    __syncthreads();
    int n = blockIdx.x * blockDim.x + tid;
    if (n < N_NODES) {
        float d = fmaxf(denom[n], 1e-16f);
        const float* s = P + (size_t)n * ldP + 3 * DH;
#pragma unroll
        for (int j = 0; j < DH; ++j) {
            float val = agg[(size_t)n * DH + j] / d + s[j];
            t[(size_t)n * DH + j] = val;
            atomicAdd(&ssum[j], val);
            atomicAdd(&ssq[j], val * val);
        }
    }
    __syncthreads();
    if (tid < DH) atomicAdd(&colstats[tid], ssum[tid]);
    else if (tid < 2 * DH) atomicAdd(&colstats[tid], ssq[tid - DH]);
}

__global__ void k_bn_lrelu(const float* __restrict__ t, const float* __restrict__ colstats,
                           const float* __restrict__ g, const float* __restrict__ b,
                           float* __restrict__ out) {
    int n = blockIdx.x * blockDim.x + threadIdx.x;
    if (n >= N_NODES) return;
    const float invN = 1.f / (float)N_NODES;
#pragma unroll
    for (int j = 0; j < DH; ++j) {
        float mu  = colstats[j] * invN;
        float var = colstats[DH + j] * invN - mu * mu;
        float y   = g[j] * (t[(size_t)n * DH + j] - mu) * rsqrtf(var + BN_EPS) + b[j];
        out[(size_t)n * DH + j] = (y >= 0.f) ? y : SLOPE * y;
    }
}

__global__ void k_concat(const float* __restrict__ a, const float* __restrict__ b,
                         float* __restrict__ out) {
    int i = blockIdx.x * blockDim.x + threadIdx.x;
    if (i >= N_NODES * DH) return;
    int n = i / DH, j = i % DH;
    out[(size_t)n * 2 * DH + j]      = a[i];
    out[(size_t)n * 2 * DH + DH + j] = b[i];
}

__global__ void k_fc(const float* __restrict__ h, const float* __restrict__ W,
                     const float* __restrict__ bias, float* __restrict__ out) {
    int n = blockIdx.x * blockDim.x + threadIdx.x;
    if (n >= N_NODES) return;
    float o0 = bias[0], o1 = bias[1];
#pragma unroll
    for (int l = 0; l < 5; ++l) {
        const float* hp = h + (size_t)l * N_NODES * DH + (size_t)n * DH;
#pragma unroll
        for (int j = 0; j < DH; ++j) {
            float x = hp[j];
            o0 += x * W[(l * DH + j) * 2 + 0];
            o1 += x * W[(l * DH + j) * 2 + 1];
        }
    }
    out[n * 2 + 0] = o0;
    out[n * 2 + 1] = o1;
}

// ---------------------------------------------------------------------------
extern "C" void kernel_launch(void* const* d_in, const int* in_sizes, int n_in,
                              void* d_out, int out_size, void* d_ws, size_t ws_size,
                              hipStream_t stream) {
    (void)in_sizes; (void)n_in; (void)out_size; (void)ws_size;

    const float* features = (const float*)d_in[0];
    const int*   edge_idx = (const int*)d_in[3];
    const int*   same2    = (const int*)d_in[4];
    const int*   diff2    = (const int*)d_in[5];
    const float* c1_W0 = (const float*)d_in[6];
    const float* c1_b0 = (const float*)d_in[7];
    const float* c2_W0 = (const float*)d_in[8];
    const float* c2_b0 = (const float*)d_in[9];
    const float* c1_W  = (const float*)d_in[10];
    const float* c1_b  = (const float*)d_in[11];
    const float* c2_W  = (const float*)d_in[12];
    const float* c2_b  = (const float*)d_in[13];
    const float* c3_W  = (const float*)d_in[14];
    const float* c3_b  = (const float*)d_in[15];
    const float* bn1_g = (const float*)d_in[16];
    const float* bn1_b = (const float*)d_in[17];
    const float* bn2_g = (const float*)d_in[18];
    const float* bn2_b = (const float*)d_in[19];
    const float* bn3_g = (const float*)d_in[20];
    const float* bn3_b = (const float*)d_in[21];
    const float* fc_W  = (const float*)d_in[22];
    const float* fc_b  = (const float*)d_in[23];
    float* out = (float*)d_out;

    // ---- workspace carve (~122 MB total) ----
    char*  ws  = (char*)d_ws;
    size_t off = 0;
    auto carve = [&](size_t bytes) -> char* {
        char* p = ws + off;
        off = (off + bytes + 255) & ~(size_t)255;
        return p;
    };
    _Float16* Wp0    = (_Float16*)carve((size_t)63 * 20 * 512 * 2);
    float*    biasp0 = (float*)carve(320 * 4);
    _Float16* WpS    = (_Float16*)carve((size_t)13 * 8192 * 2);
    float*    biaspS = (float*)carve((size_t)13 * 128 * 4);
    float*    Pbuf   = (float*)carve((size_t)N_NODES * 320 * 4);
    float*    logits = (float*)carve((size_t)N_EDGES * 4);
    int*      mInt   = (int*)carve((size_t)N_NODES * 4);
    float*    denom  = (float*)carve((size_t)N_NODES * 4);
    float*    agg    = (float*)carve((size_t)N_NODES * DH * 4);
    float*    tbuf   = (float*)carve((size_t)N_NODES * DH * 4);
    float*    cstat  = (float*)carve(2 * DH * 4);
    float*    x1     = (float*)carve((size_t)N_NODES * DH * 4);
    float*    x2     = (float*)carve((size_t)N_NODES * DH * 4);
    float*    x3     = (float*)carve((size_t)N_NODES * 2 * DH * 4);
    float*    x4     = (float*)carve((size_t)N_NODES * 2 * DH * 4);
    float*    hbuf   = (float*)carve((size_t)5 * N_NODES * DH * 4);

    const int gridE   = (N_EDGES + 255) / 256;
    const int gridN   = (N_NODES + 255) / 256;
    const int gridN20 = (N_NODES * DH + 255) / 256;
    const int gridM   = (N_NODES + 31) / 32;

    // ---- pre-pack weights into WMMA B-fragment layout (f16) ----
    pack_big<<<1024, 256, 0, stream>>>(c1_W0, c2_W0, c1_b0, c2_b0, Wp0, biasp0);
    for (int i = 0; i < 4; ++i)
        pack_small<<<32, 256, 0, stream>>>(c1_W + (size_t)i * 3200, c1_b + i * 80,
                                           WpS + (size_t)i * 8192, biaspS + i * 128);
    for (int i = 0; i < 4; ++i)
        pack_small<<<32, 256, 0, stream>>>(c2_W + (size_t)i * 3200, c2_b + i * 80,
                                           WpS + (size_t)(4 + i) * 8192,
                                           biaspS + (4 + i) * 128);
    for (int i = 0; i < 5; ++i)
        pack_small<<<32, 256, 0, stream>>>(c3_W + (size_t)i * 3200, c3_b + i * 80,
                                           WpS + (size_t)(8 + i) * 8192,
                                           biaspS + (8 + i) * 128);

    auto tconv = [&](const int* ei, const float* P, int ldP,
                     const float* g, const float* b, float* xout) {
        k_attn_init<<<gridN20, 256, 0, stream>>>(mInt, denom, agg, cstat);
        k_logits<<<gridE, 256, 0, stream>>>(ei, P, ldP, logits, mInt);
        k_expsum<<<gridE, 256, 0, stream>>>(ei, P, ldP, logits, mInt, denom, agg);
        k_finalize_stats<<<gridN, 256, 0, stream>>>(P, ldP, denom, agg, tbuf, cstat);
        k_bn_lrelu<<<gridN, 256, 0, stream>>>(tbuf, cstat, g, b, xout);
    };

    // ---- layer 0: fused big GEMM (HBM-bound on features: ~400 MB) ----
    gemm_wmma<20, 5><<<gridM, 256, 0, stream>>>(features, N_NODES, DIN, DIN,
                                                63, 320, Wp0, biasp0, Pbuf);
    tconv(same2, Pbuf,       320, bn1_g, bn1_b, x1);
    tconv(diff2, Pbuf + 160, 320, bn2_g, bn2_b, x2);

    float* Pa = Pbuf;
    float* Pb = Pbuf + (size_t)N_NODES * 80;
    float* Pc = Pbuf + (size_t)N_NODES * 160;

    k_concat<<<gridN20, 256, 0, stream>>>(x1, x2, x3);
    gemm_wmma<8, 2><<<gridM, 256, 0, stream>>>(x3, N_NODES, 2 * DH, 2 * DH, 2, 80,
                                               WpS + (size_t)8 * 8192,
                                               biaspS + 8 * 128, Pc);
    tconv(edge_idx, Pc, 80, bn3_g, bn3_b, hbuf);

    for (int l = 0; l < 4; ++l) {
        k_concat<<<gridN20, 256, 0, stream>>>(x1, x2, x3);
        k_concat<<<gridN20, 256, 0, stream>>>(x2, x1, x4);

        gemm_wmma<8, 2><<<gridM, 256, 0, stream>>>(x3, N_NODES, 2 * DH, 2 * DH, 2, 80,
                                                   WpS + (size_t)l * 8192,
                                                   biaspS + l * 128, Pa);
        tconv(same2, Pa, 80, bn1_g + (l + 1) * DH, bn1_b + (l + 1) * DH, x1);

        gemm_wmma<8, 2><<<gridM, 256, 0, stream>>>(x4, N_NODES, 2 * DH, 2 * DH, 2, 80,
                                                   WpS + (size_t)(4 + l) * 8192,
                                                   biaspS + (4 + l) * 128, Pb);
        tconv(diff2, Pb, 80, bn2_g + (l + 1) * DH, bn2_b + (l + 1) * DH, x2);

        k_concat<<<gridN20, 256, 0, stream>>>(x1, x2, x3);
        gemm_wmma<8, 2><<<gridM, 256, 0, stream>>>(x3, N_NODES, 2 * DH, 2 * DH, 2, 80,
                                                   WpS + (size_t)(9 + l) * 8192,
                                                   biaspS + (9 + l) * 128, Pc);
        tconv(edge_idx, Pc, 80, bn3_g + (l + 1) * DH, bn3_b + (l + 1) * DH,
              hbuf + (size_t)(l + 1) * N_NODES * DH);
    }

    k_fc<<<gridN, 256, 0, stream>>>(hbuf, fc_W, fc_b, out);
}
